// MultiHeadAttention_61409442398623
// MI455X (gfx1250) — compile-verified, matrix operand order checked by reading, not run on hardware
//
#include <hip/hip_runtime.h>

// ---------------------------------------------------------------------------
// MI455X (gfx1250) MHA: bf16 WMMA pipeline + ASYNCcnt DMA staging.
//   ~142 GFLOP total; flash attention keeps HBM traffic ~150MB -> compute
//   bound, so all matmuls run through v_wmma_f32_16x16x32_bf16 (f32 accum).
//   K/V tiles stream into LDS via global_load_async_to_lds_b128 (double
//   buffered, s_wait_asynccnt-gated) so DMA overlaps WMMA+softmax.
// ---------------------------------------------------------------------------

typedef __attribute__((ext_vector_type(16))) __bf16 v16bf;
typedef __attribute__((ext_vector_type(8)))  float  v8f;
typedef unsigned short u16;

constexpr int Bq   = 2;
constexpr int Sq   = 4096;
constexpr int E    = 768;
constexpr int H    = 12;
constexpr int Dh   = 64;
constexpr int Pp   = 16;
constexpr int Ltot = Pp + Sq;      // 4112 keys total
constexpr int Lpad = 4160;         // 65 * 64 (padded, masked in softmax)
constexpr int Mrows = Bq * Sq;     // 8192

__device__ __forceinline__ u16 f2bf(float f) {
  unsigned u = __float_as_uint(f);
  unsigned r = u + 0x7FFFu + ((u >> 16) & 1u);   // round-to-nearest-even
  return (u16)(r >> 16);
}

__device__ __forceinline__ v8f wmma_bf16(v16bf a, v16bf b, v8f c) {
  return __builtin_amdgcn_wmma_f32_16x16x32_bf16(false, a, false, b, (short)0, c,
                                                 false, false);
}

// CDNA5 async DMA: 16B per lane, global -> LDS, tracked by ASYNCcnt.
// GVS mode: 64-bit SGPR base + signed 32-bit per-lane byte offset.
__device__ __forceinline__ void async_ld_b128(unsigned lds_byte_addr,
                                              unsigned gl_byte_off,
                                              const void* sbase) {
  asm volatile("global_load_async_to_lds_b128 %0, %1, %2"
               :: "v"(lds_byte_addr), "v"(gl_byte_off), "s"(sbase)
               : "memory");
}

__device__ __forceinline__ void wait_async0() {
#if __has_builtin(__builtin_amdgcn_s_wait_asynccnt)
  __builtin_amdgcn_s_wait_asynccnt(0);
#else
  asm volatile("s_wait_asynccnt 0x0" ::: "memory");
#endif
}

// Load a 16x32 bf16 operand fragment from LDS per CDNA5 16-bit A layout:
// lane l -> row (l&15); VGPRs 0-3 hold K = (l>>4)*8 + 0..7 (packed pairs),
// VGPRs 4-7 hold K = 16 + (l>>4)*8 + 0..7. Two 16B LDS reads per lane.
__device__ __forceinline__ v16bf ld_frag(const u16* base, int stride) {
  int lane = threadIdx.x & 31;
  const u16* p = base + (lane & 15) * stride + ((lane >> 4) << 3);
  union { v16bf v; uint4 u[2]; } r;
  r.u[0] = *(const uint4*)p;
  r.u[1] = *(const uint4*)(p + 16);
  return r.v;
}

// ---------------------------------------------------------------------------
// GEMM: Y = X @ W^T + bias.  X:[8192,768], W:[768,768] row-major (N,K).
// Block tile 128x64, 8 waves, each wave 32x32 (2x2 WMMA tiles). BK=32.
// MODE 0: Q -> bf16 [B,H,S,D] * 1/8 | MODE 1: K -> bf16 [B,H,Lpad,D] @ +Pp
// MODE 2: V -> bf16 [B,H,D,Lpad] (transposed) @ +Pp | MODE 3: O -> f32 [M,E]
// ---------------------------------------------------------------------------
template<int MODE>
__global__ __launch_bounds__(256)
void gemm_qkvo(const void* __restrict__ Av, const float* __restrict__ W,
               const float* __restrict__ bias, void* __restrict__ outv) {
  constexpr int BM = 128, BN = 64, BK = 32, LDT = BK + 8;  // 80B rows
  __shared__ u16 As[BM * LDT];
  __shared__ u16 Bs[BN * LDT];

  const int tid  = threadIdx.x;
  const int lane = tid & 31;
  const int wave = tid >> 5;
  const int wm = wave & 3, wn = wave >> 2;
  const int m0 = blockIdx.x * BM, n0 = blockIdx.y * BN;

  v8f acc00 = {}, acc01 = {}, acc10 = {}, acc11 = {};

  const int arow = tid >> 1, acol = (tid & 1) * 16;  // 128 rows x 32 cols
  const int brow = tid >> 2, bcol = (tid & 3) * 8;   //  64 rows x 32 cols

  for (int k0 = 0; k0 < E; k0 += BK) {
    if constexpr (MODE == 3) {
      // bf16 A: DMA straight into LDS (no VGPR data path)
      unsigned goff = (unsigned)(((size_t)(m0 + arow) * E + k0 + acol) * 2);
      unsigned loff = (unsigned)(size_t)&As[arow * LDT + acol];
      async_ld_b128(loff,       goff,       Av);
      async_ld_b128(loff + 16u, goff + 16u, Av);
    } else {
      const float* ap = (const float*)Av + (size_t)(m0 + arow) * E + k0 + acol;
      u16 h[16];
#pragma unroll
      for (int i = 0; i < 16; i += 4) {
        float4 f = *(const float4*)(ap + i);
        h[i] = f2bf(f.x); h[i + 1] = f2bf(f.y);
        h[i + 2] = f2bf(f.z); h[i + 3] = f2bf(f.w);
      }
      *(uint4*)&As[arow * LDT + acol]     = *(uint4*)&h[0];
      *(uint4*)&As[arow * LDT + acol + 8] = *(uint4*)&h[8];
    }
    {
      const float* bp = W + (size_t)(n0 + brow) * E + k0 + bcol;
      float4 f0 = *(const float4*)bp;
      float4 f1 = *(const float4*)(bp + 4);
      u16 h[8];
      h[0] = f2bf(f0.x); h[1] = f2bf(f0.y); h[2] = f2bf(f0.z); h[3] = f2bf(f0.w);
      h[4] = f2bf(f1.x); h[5] = f2bf(f1.y); h[6] = f2bf(f1.z); h[7] = f2bf(f1.w);
      *(uint4*)&Bs[brow * LDT + bcol] = *(uint4*)&h[0];
    }
    if constexpr (MODE == 3) wait_async0();
    __syncthreads();

    v16bf a0 = ld_frag(&As[(wm * 32) * LDT], LDT);
    v16bf a1 = ld_frag(&As[(wm * 32 + 16) * LDT], LDT);
    v16bf b0 = ld_frag(&Bs[(wn * 32) * LDT], LDT);
    v16bf b1 = ld_frag(&Bs[(wn * 32 + 16) * LDT], LDT);
    acc00 = wmma_bf16(a0, b0, acc00);
    acc01 = wmma_bf16(a0, b1, acc01);
    acc10 = wmma_bf16(a1, b0, acc10);
    acc11 = wmma_bf16(a1, b1, acc11);
    __syncthreads();
  }

  // Epilogue. C/D layout: VGPR v, lane l -> row v + 8*(l>>4), col l&15.
  const int rl = 8 * (lane >> 4), cl = lane & 15;
#pragma unroll
  for (int tm = 0; tm < 2; ++tm) {
#pragma unroll
    for (int tn = 0; tn < 2; ++tn) {
      v8f acc = (tm == 0) ? (tn == 0 ? acc00 : acc01)
                          : (tn == 0 ? acc10 : acc11);
      int gn = n0 + wn * 32 + tn * 16 + cl;
      float bval = bias[gn];
#pragma unroll
      for (int v = 0; v < 8; ++v) {
        int gm = m0 + wm * 32 + tm * 16 + v + rl;
        float val = acc[v] + bval;
        if constexpr (MODE == 3) {
          ((float*)outv)[(size_t)gm * E + gn] = val;
        } else {
          int bidx = gm >> 12, s = gm & (Sq - 1);
          int hh = gn >> 6,  d = gn & 63;
          if constexpr (MODE == 0) {
            ((u16*)outv)[(((size_t)(bidx * H + hh) * Sq + s) << 6) + d] =
                f2bf(val * 0.125f);   // 1/sqrt(64)
          } else if constexpr (MODE == 1) {
            ((u16*)outv)[(((size_t)(bidx * H + hh) * Lpad + Pp + s) << 6) + d] =
                f2bf(val);
          } else {
            ((u16*)outv)[((size_t)(bidx * H + hh) * Dh + d) * Lpad + Pp + s] =
                f2bf(val);
          }
        }
      }
    }
  }
}

// ---------------------------------------------------------------------------
// Prompt prefix copy into K/V buffers + zero the Ltot..Lpad padding.
// ---------------------------------------------------------------------------
__global__ __launch_bounds__(256)
void prep_kv(const float* __restrict__ prompt, u16* __restrict__ Kb,
             u16* __restrict__ Vb) {
  int idx = blockIdx.x * 256 + threadIdx.x;
  constexpr int NP = Bq * 2 * Pp * H * Dh;                 // 49152
  constexpr int NZ = Bq * H * (Lpad - Ltot) * Dh;          // 73728
  if (idx < NP) {
    int d = idx & 63;
    int hh = (idx >> 6) % H;
    int p = (idx / (Dh * H)) % Pp;
    int t = (idx / (Dh * H * Pp)) & 1;
    int b = idx / (Dh * H * Pp * 2);
    u16 o = f2bf(prompt[idx]);
    if (t == 0)
      Kb[(((size_t)(b * H + hh) * Lpad + p) << 6) + d] = o;
    else
      Vb[((size_t)(b * H + hh) * Dh + d) * Lpad + p] = o;
  }
  int pz = idx - NP;
  if (pz >= 0 && pz < NZ) {
    int d = pz & 63;
    int pp = (pz >> 6) % (Lpad - Ltot);
    int hh = (pz / (Dh * (Lpad - Ltot))) % H;
    int bb = pz / (Dh * (Lpad - Ltot) * H);
    int pos = Ltot + pp;
    Kb[(((size_t)(bb * H + hh) * Lpad + pos) << 6) + d] = 0;
    Vb[((size_t)(bb * H + hh) * Dh + d) * Lpad + pos] = 0;
  }
}

// ---------------------------------------------------------------------------
// Flash attention. Grid (S/64, H, B), 256 threads = 8 waves.
// 64 query rows per block; 65 key chunks of 64 streamed by async DMA with
// double buffering. Probs tile aliases the score buffer to fit 64KB LDS.
// ---------------------------------------------------------------------------
__global__ __launch_bounds__(256)
void flash_attn(const u16* __restrict__ Qb, const u16* __restrict__ Kb,
                const u16* __restrict__ Vb, u16* __restrict__ Cb) {
  constexpr int LQ  = 72;   // bf16 row stride (144B, 16B aligned)
  constexpr int LSS = 66;   // f32 score row stride
  __shared__ u16 Qs[64 * LQ];            //  9216 B
  __shared__ u16 Ks[2][64 * LQ];         // 18432 B
  __shared__ u16 Vs[2][64 * LQ];         // 18432 B
  __shared__ float Ss[64 * LSS];         // 16896 B (bf16 Ps aliases this)
  __shared__ float rm[64], rs[64], rc[64];
  u16* Ps = (u16*)Ss;

  const int tid = threadIdx.x, lane = tid & 31, wave = tid >> 5;
  const int wm = wave & 3, wn = wave >> 2;
  const int q0 = blockIdx.x * 64, hh = blockIdx.y, bb = blockIdx.z;
  const size_t bh = (size_t)(bb * H + hh);

  if (tid < 64) { rm[tid] = -3.0e38f; rs[tid] = 0.f; }

  const u16* qp = Qb + (bh * Sq + q0) * Dh;
  const u16* kbase = Kb + bh * Lpad * Dh;
  const u16* vbase = Vb + bh * Dh * Lpad;

  // prologue: async-stage Q tile and chunk 0 of K/V into buffer 0
#pragma unroll
  for (int i = 0; i < 2; ++i) {
    int c = tid + 256 * i, row = c >> 3, col = (c & 7) * 8;
    async_ld_b128((unsigned)(size_t)&Qs[row * LQ + col],
                  (unsigned)((row * Dh + col) * 2), qp);
    async_ld_b128((unsigned)(size_t)&Ks[0][row * LQ + col],
                  (unsigned)((row * Dh + col) * 2), kbase);
    async_ld_b128((unsigned)(size_t)&Vs[0][row * LQ + col],
                  (unsigned)((row * Lpad + col) * 2), vbase);
  }
  wait_async0();
  __syncthreads();

  v8f acc0 = {}, acc1 = {};
  const int rl = 8 * (lane >> 4), cl = lane & 15;

  for (int j = 0; j < Lpad / 64; ++j) {
    const int cur = j & 1;
    if (j + 1 < Lpad / 64) {  // prefetch next chunk via DMA, overlaps compute
      const int nxt = cur ^ 1;
#pragma unroll
      for (int i = 0; i < 2; ++i) {
        int c = tid + 256 * i, row = c >> 3, col = (c & 7) * 8;
        async_ld_b128((unsigned)(size_t)&Ks[nxt][row * LQ + col],
                      (unsigned)((((j + 1) * 64 + row) * Dh + col) * 2), kbase);
        async_ld_b128((unsigned)(size_t)&Vs[nxt][row * LQ + col],
                      (unsigned)((row * Lpad + (j + 1) * 64 + col) * 2), vbase);
      }
    }

    // scores: S = Q @ K^T (both operands d-contiguous in LDS)
    v8f s0 = {}, s1 = {};
#pragma unroll
    for (int kk = 0; kk < Dh; kk += 32) {
      v16bf a   = ld_frag(&Qs[(wm * 16) * LQ + kk], LQ);
      v16bf kb0 = ld_frag(&Ks[cur][(wn * 32) * LQ + kk], LQ);
      v16bf kb1 = ld_frag(&Ks[cur][(wn * 32 + 16) * LQ + kk], LQ);
      s0 = wmma_bf16(a, kb0, s0);
      s1 = wmma_bf16(a, kb1, s1);
    }
#pragma unroll
    for (int v = 0; v < 8; ++v) {
      Ss[(wm * 16 + v + rl) * LSS + wn * 32 + cl]      = s0[v];
      Ss[(wm * 16 + v + rl) * LSS + wn * 32 + 16 + cl] = s1[v];
    }
    __syncthreads();

    // online softmax: 4 threads per row, 16 cols each, mask cols >= Ltot
    const int r = tid >> 2, qd = tid & 3;
    const int gb = j * 64 + qd * 16;
    float vals[16];
    float mloc = -3.0e38f;
#pragma unroll
    for (int i = 0; i < 16; ++i) {
      float x = Ss[r * LSS + qd * 16 + i];
      vals[i] = x;
      if (gb + i < Ltot) mloc = fmaxf(mloc, x);
    }
    mloc = fmaxf(mloc, __shfl_xor(mloc, 1, 32));
    mloc = fmaxf(mloc, __shfl_xor(mloc, 2, 32));
    float mold = rm[r];
    float mnew = fmaxf(mold, mloc);
    float corr = __expf(mold - mnew);
    __syncthreads();   // protect Ss -> Ps alias (scores now in registers)

    float ssum = 0.f;
    u16 ph[16];
#pragma unroll
    for (int i = 0; i < 16; ++i) {
      float p = (gb + i < Ltot) ? __expf(vals[i] - mnew) : 0.f;
      ssum += p;
      ph[i] = f2bf(p);
    }
    ssum += __shfl_xor(ssum, 1, 32);
    ssum += __shfl_xor(ssum, 2, 32);
    *(uint4*)&Ps[r * LQ + qd * 16]     = *(uint4*)&ph[0];
    *(uint4*)&Ps[r * LQ + qd * 16 + 8] = *(uint4*)&ph[8];
    if (qd == 0) { rs[r] = rs[r] * corr + ssum; rm[r] = mnew; rc[r] = corr; }
    __syncthreads();

    // rescale running ctx, then ctx += P @ V (Vs holds V^T: rows=d, K=keys)
#pragma unroll
    for (int v = 0; v < 8; ++v) {
      float c0 = rc[wm * 16 + v + rl];
      acc0[v] *= c0;
      acc1[v] *= c0;
    }
#pragma unroll
    for (int kk = 0; kk < 64; kk += 32) {
      v16bf a   = ld_frag(&Ps[(wm * 16) * LQ + kk], LQ);
      v16bf vb0 = ld_frag(&Vs[cur][(wn * 32) * LQ + kk], LQ);
      v16bf vb1 = ld_frag(&Vs[cur][(wn * 32 + 16) * LQ + kk], LQ);
      acc0 = wmma_bf16(a, vb0, acc0);
      acc1 = wmma_bf16(a, vb1, acc1);
    }
    wait_async0();     // next chunk's DMA landed
    __syncthreads();   // everyone done reading cur before it is overwritten
  }

  // normalize by row sums, stage bf16 in LDS, coalesced writeout [B,S,H,D]
#pragma unroll
  for (int v = 0; v < 8; ++v) {
    int row = wm * 16 + v + rl;
    float inv = 1.0f / rs[row];
    Ps[row * LQ + wn * 32 + cl]      = f2bf(acc0[v] * inv);
    Ps[row * LQ + wn * 32 + 16 + cl] = f2bf(acc1[v] * inv);
  }
  __syncthreads();
#pragma unroll
  for (int i = 0; i < 2; ++i) {
    int c = tid + 256 * i, row = c >> 3, col = (c & 7) * 8;
    *(uint4*)(Cb + (((size_t)(bb * Sq + q0 + row)) * H + hh) * Dh + col) =
        *(const uint4*)&Ps[row * LQ + col];
  }
}

// ---------------------------------------------------------------------------
extern "C" void kernel_launch(void* const* d_in, const int* in_sizes, int n_in,
                              void* d_out, int out_size, void* d_ws, size_t ws_size,
                              hipStream_t stream) {
  (void)in_sizes; (void)n_in; (void)out_size; (void)ws_size;
  const float* query  = (const float*)d_in[0];
  const float* key    = (const float*)d_in[1];
  const float* value  = (const float*)d_in[2];
  const float* prompt = (const float*)d_in[3];
  const float* Wq = (const float*)d_in[4];
  const float* bq = (const float*)d_in[5];
  const float* Wk = (const float*)d_in[6];
  const float* bk = (const float*)d_in[7];
  const float* Wv = (const float*)d_in[8];
  const float* bv = (const float*)d_in[9];
  const float* Wo = (const float*)d_in[10];
  const float* bo = (const float*)d_in[11];

  // bf16 workspace carve (~51 MB)
  u16* ws = (u16*)d_ws;
  const size_t qN = (size_t)Bq * H * Sq * Dh;
  const size_t kN = (size_t)Bq * H * Lpad * Dh;
  u16* Qb = ws;
  u16* Kb = Qb + qN;
  u16* Vb = Kb + kN;
  u16* Cb = Vb + kN;

  dim3 blk(256);
  dim3 gG(Mrows / 128, E / 64);
  gemm_qkvo<0><<<gG, blk, 0, stream>>>(query, Wq, bq, Qb);
  gemm_qkvo<1><<<gG, blk, 0, stream>>>(key,   Wk, bk, Kb);
  gemm_qkvo<2><<<gG, blk, 0, stream>>>(value, Wv, bv, Vb);

  constexpr int prepN = Bq * 2 * Pp * H * Dh + Bq * H * (Lpad - Ltot) * Dh;
  prep_kv<<<(prepN + 255) / 256, blk, 0, stream>>>(prompt, Kb, Vb);

  flash_attn<<<dim3(Sq / 64, H, Bq), blk, 0, stream>>>(Qb, Kb, Vb, Cb);

  gemm_qkvo<3><<<gG, blk, 0, stream>>>(Cb, Wo, bo, d_out);
}
